// CausalSelfAttention_75033078662000
// MI455X (gfx1250) — matrix operand order repaired and checked by reading, not compiled, not run on hardware
//
#include <hip/hip_runtime.h>

#define D_MODEL 1024
#define N_HEADS 16
#define HEAD_DIM 64
#define SEQ_LEN 2048
#define BATCH 4
#define NTOK (BATCH * SEQ_LEN) // 8192

typedef __bf16 bf16;
typedef __attribute__((ext_vector_type(8)))  __bf16 bf16x8;
typedef __attribute__((ext_vector_type(16))) __bf16 bf16x16;
typedef __attribute__((ext_vector_type(8)))  float  f32x8;

__device__ __forceinline__ f32x8 wmma_bf16(bf16x16 a, bf16x16 b, f32x8 c) {
    return __builtin_amdgcn_wmma_f32_16x16x32_bf16(
        /*neg_a=*/false, a, /*neg_b=*/false, b,
        /*c_mod=*/(short)0, c, /*reuse_a=*/false, /*reuse_b=*/false);
}

// A-fragment (16x32 bf16, M=rows): lane = {half,m}; row m, K = half*16 + 0..15
__device__ __forceinline__ bf16x16 load_a_frag(const bf16* base, int lda, int lane) {
    const int m = lane & 15, half = lane >> 4;
    const bf16* p = base + (size_t)m * lda + half * 16;
    bf16x8 lo = *(const bf16x8*)(p);
    bf16x8 hi = *(const bf16x8*)(p + 8);
    return __builtin_shufflevector(lo, hi, 0, 1, 2, 3, 4, 5, 6, 7,
                                   8, 9, 10, 11, 12, 13, 14, 15);
}

// B-fragment (32x16 bf16, conceptual B[k][n]) from transposed storage Bt[n][k]:
// lane = {half,n}; column n, K = half*16 + 0..15 (contiguous 32B)
__device__ __forceinline__ bf16x16 load_b_frag_t(const bf16* baseT, int ldt, int lane) {
    const int n = lane & 15, half = lane >> 4;
    return *(const bf16x16*)(baseT + (size_t)n * ldt + half * 16);
}

// Async copy 16B global -> LDS (tracked by ASYNCcnt). LDS operand is the
// 32-bit LDS byte offset (low 32 bits of the generic pointer to __shared__).
__device__ __forceinline__ void async_b128(const bf16* g, const bf16* lds) {
    unsigned off = (unsigned)(size_t)lds;
    asm volatile("global_load_async_to_lds_b128 %0, %1, off"
                 :: "v"(off), "v"(g)
                 : "memory");
}

template <int N>
__device__ __forceinline__ void wait_asynccnt() {
    asm volatile("s_wait_asynccnt %0" :: "i"(N) : "memory");
}

// ---------------- prepack: f32 -> bf16, weights transposed to [N][K] ----------------
__global__ void prepack_kernel(const float* __restrict__ x,
                               const float* __restrict__ wqkv,
                               const float* __restrict__ wproj,
                               bf16* __restrict__ xb,
                               bf16* __restrict__ wqkvT,
                               bf16* __restrict__ wprojT) {
    size_t i = (size_t)blockIdx.x * blockDim.x + threadIdx.x;
    const size_t NX = (size_t)NTOK * D_MODEL;
    if (i < NX) xb[i] = (bf16)x[i];
    if (i < (size_t)3 * D_MODEL * D_MODEL) {
        int n = (int)(i >> 10), kk = (int)(i & 1023);
        wqkvT[i] = (bf16)wqkv[(size_t)kk * (3 * D_MODEL) + n];
    }
    if (i < (size_t)D_MODEL * D_MODEL) {
        int n = (int)(i >> 10), kk = (int)(i & 1023);
        wprojT[i] = (bf16)wproj[(size_t)kk * D_MODEL + n];
    }
}

// Double-buffered async-LDS GEMM core: acc[4] = A[m0+wave*16 .. +16][:] x BT^T[n0 .. n0+64]
__device__ __forceinline__ void gemm_core_async(const bf16* __restrict__ A,
                                                const bf16* __restrict__ BT,
                                                int m0, int n0, int tid,
                                                bf16 (*At)[128 * 32],
                                                bf16 (*Bt)[64 * 32],
                                                f32x8 acc[4]) {
    const int lane = tid & 31, wave = tid >> 5;
    // A tile: 128x32 bf16 = 512 x 16B chunks -> 2 per thread
    const int ca0 = tid, ca1 = tid + 256;
    const int ar0 = ca0 >> 2, ac0 = (ca0 & 3) * 8;
    const int ar1 = ca1 >> 2, ac1 = (ca1 & 3) * 8;
    // B tile: 64x32 bf16 = 256 x 16B chunks -> 1 per thread
    const int br = tid >> 2, bc = (tid & 3) * 8;

    const bf16* gA0 = A + (size_t)(m0 + ar0) * D_MODEL + ac0;
    const bf16* gA1 = A + (size_t)(m0 + ar1) * D_MODEL + ac1;
    const bf16* gB = BT + (size_t)(n0 + br) * D_MODEL + bc;

    // Prologue: tile 0 -> buffer 0
    async_b128(gA0, &At[0][ar0 * 32 + ac0]);
    async_b128(gA1, &At[0][ar1 * 32 + ac1]);
    async_b128(gB, &Bt[0][br * 32 + bc]);

    const int NIT = D_MODEL / 32;
    for (int i = 0; i < NIT; ++i) {
        const int cur = i & 1, nxt = cur ^ 1;
        if (i + 1 < NIT) {
            const int kb = (i + 1) * 32;
            async_b128(gA0 + kb, &At[nxt][ar0 * 32 + ac0]);
            async_b128(gA1 + kb, &At[nxt][ar1 * 32 + ac1]);
            async_b128(gB + kb, &Bt[nxt][br * 32 + bc]);
            wait_asynccnt<3>(); // tile i landed (in-order), tile i+1 in flight
        } else {
            wait_asynccnt<0>();
        }
        __syncthreads();
        bf16x16 af = load_a_frag(&At[cur][wave * 16 * 32], 32, lane);
#pragma unroll
        for (int nt = 0; nt < 4; ++nt) {
            bf16x16 bfrag = load_b_frag_t(&Bt[cur][nt * 16 * 32], 32, lane);
            acc[nt] = wmma_bf16(af, bfrag, acc[nt]);
        }
        __syncthreads(); // all waves done reading buf[cur] before it is overwritten
    }
}

// ---------------- QKV GEMM: [8192,1024] x [1024,3072] -> scatter Q/K/V ----------------
__global__ __launch_bounds__(256) void qkv_gemm_kernel(const bf16* __restrict__ x,
                                                       const bf16* __restrict__ wT,
                                                       bf16* __restrict__ qws,
                                                       bf16* __restrict__ kws,
                                                       bf16* __restrict__ vws) {
    __shared__ alignas(16) bf16 At[2][128 * 32];
    __shared__ alignas(16) bf16 Bt[2][64 * 32];
    const int tid = threadIdx.x;
    const int lane = tid & 31, wave = tid >> 5;
    const int m0 = blockIdx.y * 128;
    const int n0 = blockIdx.x * 64;

    f32x8 acc[4] = {};
    gemm_core_async(x, wT, m0, n0, tid, At, Bt, acc);

    const int half = lane >> 4, nl = lane & 15;
#pragma unroll
    for (int nt = 0; nt < 4; ++nt) {
#pragma unroll
        for (int j = 0; j < 8; ++j) {
            int rt = m0 + wave * 16 + j + 8 * half; // global token row
            int col = n0 + nt * 16 + nl;            // 0..3071
            int b = rt >> 11, t = rt & (SEQ_LEN - 1);
            int sec = col >> 10, f = col & (D_MODEL - 1);
            int h = f >> 6, d = f & 63;
            int bh = b * N_HEADS + h;
            bf16 val = (bf16)acc[nt][j];
            if (sec == 0)      qws[(((size_t)bh * SEQ_LEN + t) << 6) + d] = val;
            else if (sec == 1) kws[(((size_t)bh * SEQ_LEN + t) << 6) + d] = val;
            else               vws[(((size_t)bh * 64 + d) << 11) + t] = val; // V transposed [hd][T]
        }
    }
}

// ---------------- Flash attention: 1 wave per 16-row query tile ----------------
__global__ __launch_bounds__(128) void attn_kernel(const bf16* __restrict__ q,
                                                   const bf16* __restrict__ k,
                                                   const bf16* __restrict__ v,
                                                   bf16* __restrict__ o) {
    __shared__ alignas(32) bf16 pbuf[4][16 * 32];
    const int lane = threadIdx.x & 31, wave = threadIdx.x >> 5;
    const int bh = blockIdx.y;               // 0..63
    const int qtile = blockIdx.x * 4 + wave; // 0..127
    const int q0 = qtile * 16;
    const int half = lane >> 4, nl = lane & 15;

    const bf16* qbase = q + (size_t)bh * SEQ_LEN * HEAD_DIM;
    const bf16* kbase = k + (size_t)bh * SEQ_LEN * HEAD_DIM;
    const bf16* vbase = v + (size_t)bh * HEAD_DIM * SEQ_LEN; // [64][2048]

    bf16x16 aq0 = load_a_frag(qbase + (size_t)q0 * HEAD_DIM + 0, HEAD_DIM, lane);
    bf16x16 aq1 = load_a_frag(qbase + (size_t)q0 * HEAD_DIM + 32, HEAD_DIM, lane);

    f32x8 o0 = {}, o1 = {}, o2 = {}, o3 = {};
    float mrun[8], lrun[8];
#pragma unroll
    for (int j = 0; j < 8; ++j) { mrun[j] = -1e30f; lrun[j] = 0.0f; }

    const int nkb = (q0 + 16 + 31) >> 5;
    bf16* pb = &pbuf[wave][0];

    for (int kb2 = 0; kb2 < nkb; ++kb2) {
        const int key0 = kb2 * 32;
        if (kb2 + 1 < nkb) { // prefetch next K/V tiles (lowers to global_prefetch_b8)
            __builtin_prefetch(kbase + (size_t)(key0 + 32) * 64, 0, 1);
            __builtin_prefetch(vbase + key0 + 32, 0, 1);
        }
        f32x8 s0 = {}, s1 = {};
        bf16x16 bk;
        bk = load_b_frag_t(kbase + (size_t)(key0 + 0) * 64 + 0, 64, lane);   s0 = wmma_bf16(aq0, bk, s0);
        bk = load_b_frag_t(kbase + (size_t)(key0 + 0) * 64 + 32, 64, lane);  s0 = wmma_bf16(aq1, bk, s0);
        bk = load_b_frag_t(kbase + (size_t)(key0 + 16) * 64 + 0, 64, lane);  s1 = wmma_bf16(aq0, bk, s1);
        bk = load_b_frag_t(kbase + (size_t)(key0 + 16) * 64 + 32, 64, lane); s1 = wmma_bf16(aq1, bk, s1);

        const float scale = 0.125f; // 1/sqrt(64)
#pragma unroll
        for (int j = 0; j < 8; ++j) {
            int row = q0 + j + 8 * half;
            float v0 = s0[j] * scale, v1 = s1[j] * scale;
            if (key0 + nl > row)      v0 = -1e30f;
            if (key0 + 16 + nl > row) v1 = -1e30f;
            float mx = fmaxf(v0, v1);
            mx = fmaxf(mx, __shfl_xor(mx, 1));
            mx = fmaxf(mx, __shfl_xor(mx, 2));
            mx = fmaxf(mx, __shfl_xor(mx, 4));
            mx = fmaxf(mx, __shfl_xor(mx, 8));
            float mnew = fmaxf(mrun[j], mx);
            float corr = __expf(mrun[j] - mnew);
            mrun[j] = mnew;
            float e0 = __expf(v0 - mnew);
            float e1 = __expf(v1 - mnew);
            float rs = e0 + e1;
            rs += __shfl_xor(rs, 1);
            rs += __shfl_xor(rs, 2);
            rs += __shfl_xor(rs, 4);
            rs += __shfl_xor(rs, 8);
            lrun[j] = lrun[j] * corr + rs;
            o0[j] *= corr; o1[j] *= corr; o2[j] *= corr; o3[j] *= corr;
            int prow = j + 8 * half;
            pb[prow * 32 + nl] = (bf16)e0;
            pb[prow * 32 + 16 + nl] = (bf16)e1;
        }
        asm volatile("" ::: "memory"); // keep per-wave DS store->load order (HW is in-order per wave)
        bf16x16 ap = load_a_frag(pb, 32, lane);
        bf16x16 bv;
        bv = *(const bf16x16*)(vbase + (size_t)(0 * 16 + nl) * SEQ_LEN + key0 + half * 16); o0 = wmma_bf16(ap, bv, o0);
        bv = *(const bf16x16*)(vbase + (size_t)(1 * 16 + nl) * SEQ_LEN + key0 + half * 16); o1 = wmma_bf16(ap, bv, o1);
        bv = *(const bf16x16*)(vbase + (size_t)(2 * 16 + nl) * SEQ_LEN + key0 + half * 16); o2 = wmma_bf16(ap, bv, o2);
        bv = *(const bf16x16*)(vbase + (size_t)(3 * 16 + nl) * SEQ_LEN + key0 + half * 16); o3 = wmma_bf16(ap, bv, o3);
    }

    const int b = bh >> 4, h = bh & 15;
#pragma unroll
    for (int j = 0; j < 8; ++j) {
        float inv = 1.0f / lrun[j];
        int trow = q0 + j + 8 * half;
        size_t rb = ((size_t)(b * SEQ_LEN + trow)) * D_MODEL + h * HEAD_DIM + nl;
        o[rb + 0]  = (bf16)(o0[j] * inv);
        o[rb + 16] = (bf16)(o1[j] * inv);
        o[rb + 32] = (bf16)(o2[j] * inv);
        o[rb + 48] = (bf16)(o3[j] * inv);
    }
}

// ---------------- Output projection: [8192,1024] x [1024,1024] -> f32 ----------------
__global__ __launch_bounds__(256) void proj_gemm_kernel(const bf16* __restrict__ A,
                                                        const bf16* __restrict__ wT,
                                                        float* __restrict__ out) {
    __shared__ alignas(16) bf16 At[2][128 * 32];
    __shared__ alignas(16) bf16 Bt[2][64 * 32];
    const int tid = threadIdx.x;
    const int lane = tid & 31, wave = tid >> 5;
    const int m0 = blockIdx.y * 128;
    const int n0 = blockIdx.x * 64;

    f32x8 acc[4] = {};
    gemm_core_async(A, wT, m0, n0, tid, At, Bt, acc);

    const int half = lane >> 4, nl = lane & 15;
#pragma unroll
    for (int nt = 0; nt < 4; ++nt) {
#pragma unroll
        for (int j = 0; j < 8; ++j) {
            size_t r = (size_t)(m0 + wave * 16 + j + 8 * half);
            out[r * D_MODEL + n0 + nt * 16 + nl] = acc[nt][j];
        }
    }
}

extern "C" void kernel_launch(void* const* d_in, const int* in_sizes, int n_in,
                              void* d_out, int out_size, void* d_ws, size_t ws_size,
                              hipStream_t stream) {
    const float* x = (const float*)d_in[0];
    const float* wqkv = (const float*)d_in[1];
    const float* wproj = (const float*)d_in[2];

    char* ws = (char*)d_ws;
    bf16* xb = (bf16*)ws;     ws += (size_t)NTOK * D_MODEL * 2;        // 16 MB
    bf16* wqkvT = (bf16*)ws;  ws += (size_t)3 * D_MODEL * D_MODEL * 2; // 6 MB
    bf16* wprojT = (bf16*)ws; ws += (size_t)D_MODEL * D_MODEL * 2;     // 2 MB
    bf16* qws = (bf16*)ws;    ws += (size_t)NTOK * D_MODEL * 2;        // 16 MB
    bf16* kws = (bf16*)ws;    ws += (size_t)NTOK * D_MODEL * 2;        // 16 MB
    bf16* vws = (bf16*)ws;    ws += (size_t)NTOK * D_MODEL * 2;        // 16 MB
    bf16* ows = (bf16*)ws;    ws += (size_t)NTOK * D_MODEL * 2;        // 16 MB

    {
        size_t n = (size_t)NTOK * D_MODEL;
        int blocks = (int)((n + 255) / 256);
        prepack_kernel<<<blocks, 256, 0, stream>>>(x, wqkv, wproj, xb, wqkvT, wprojT);
    }
    {
        dim3 grid(3 * D_MODEL / 64, NTOK / 128); // 48 x 64
        qkv_gemm_kernel<<<grid, 256, 0, stream>>>(xb, wqkvT, qws, kws, vws);
    }
    {
        dim3 grid(SEQ_LEN / 16 / 4, BATCH * N_HEADS); // 32 x 64
        attn_kernel<<<grid, 128, 0, stream>>>(qws, kws, vws, ows);
    }
    {
        dim3 grid(D_MODEL / 64, NTOK / 128); // 16 x 64
        proj_gemm_kernel<<<grid, 256, 0, stream>>>(ows, wprojT, (float*)d_out);
    }
}
